// GPTBlock_39728447488625
// MI455X (gfx1250) — matrix operand order repaired and checked by reading, not compiled, not run on hardware
//
#include <hip/hip_runtime.h>

// ---------------------------------------------------------------------------
// GPT block on MI455X (gfx1250): bf16 WMMA GEMMs + f32 LN/softmax epilogues.
// Compute-bound (~0.45 TFLOP vs ~170MB traffic) -> everything through
// v_wmma_f32_16x16x32_bf16, wave32, 8 waves/WG, 128x128x32 tiles,
// async global->LDS double buffering (ASYNCcnt), causal tile-skip + K-cap.
// ---------------------------------------------------------------------------

typedef __attribute__((ext_vector_type(16))) __bf16 v16bf;
typedef __attribute__((ext_vector_type(8)))  __bf16 bf16x8;
typedef __attribute__((ext_vector_type(4)))  __bf16 bf16x4;
typedef __attribute__((ext_vector_type(8)))  float  v8f;
typedef __attribute__((ext_vector_type(4)))  int    v4i;

#if __has_builtin(__builtin_amdgcn_global_load_async_to_lds_b128)
#define HAVE_ASYNC_LDS 1
#else
#define HAVE_ASYNC_LDS 0
#endif

static __device__ __forceinline__ void wait_asynccnt0() {
#if __has_builtin(__builtin_amdgcn_s_wait_asynccnt)
  __builtin_amdgcn_s_wait_asynccnt(0);
#else
  asm volatile("s_wait_asynccnt 0x0" ::: "memory");
#endif
}

#if HAVE_ASYNC_LDS
typedef __attribute__((address_space(1))) v4i* as1_v4i_ptr;
typedef __attribute__((address_space(3))) v4i* as3_v4i_ptr;
static __device__ __forceinline__ void async_cp16(const __bf16* g, __bf16* l) {
  __builtin_amdgcn_global_load_async_to_lds_b128(
      (as1_v4i_ptr)g, (as3_v4i_ptr)l, 0, 0);
}
#endif

static __device__ __forceinline__ __bf16 f2bf(float f) {
  union { float f; unsigned u; } v; v.f = f;
  unsigned r = v.u + 0x7fffu + ((v.u >> 16) & 1u);   // round-to-nearest-even
  unsigned short h = (unsigned short)(r >> 16);
  __bf16 b; __builtin_memcpy(&b, &h, 2);
  return b;
}

static __device__ __forceinline__ bf16x8 ld8(const __bf16* p) {
  return *(const bf16x8*)p;
}

// ---------------------------------------------------------------------------
// Tiled NT GEMM:  C[M,N] = act(scale * A[M,K] * B[N,K]^T + bias) + resid
// A,B bf16 row-major.  blockDim=256 (8 waves), tile BM=BN=128, BK=32.
// Wave grid 2(M)x4(N): each wave owns 64x32 = 4x2 WMMA 16x16 accumulators.
// ---------------------------------------------------------------------------
template<bool OUT_BF16, bool TRANS_OUT, bool RELU, bool CAUSAL_SKIP, bool KCAP,
         bool HAS_BIAS, bool HAS_RES>
__global__ __launch_bounds__(256)
void gemm_nt(const __bf16* __restrict__ A, int lda, long sA,
             const __bf16* __restrict__ Bm, int ldb, long sB,
             void* Cout, int ldc, long sC,
             const float* __restrict__ bias,
             const float* resid,
             float scale, int M, int N, int K)
{
  const int n0 = blockIdx.x * 128;
  const int m0 = blockIdx.y * 128;
  if (CAUSAL_SKIP && n0 > m0 + 127) return;   // fully-masked score tile

  const long bz = blockIdx.z;
  A  += (size_t)bz * sA;
  Bm += (size_t)bz * sB;

  __shared__ __bf16 As[2][128][40];   // 40-elem row pad: conflict-light b128
  __shared__ __bf16 Bs[2][128][40];

  const int tid   = threadIdx.x;
  const int srow  = tid >> 1;          // staging: 2 threads per row
  const int skoff = (tid & 1) * 16;    // each thread: 16 contiguous bf16

  const int lane  = tid & 31;
  const int wave  = tid >> 5;
  const int wm    = wave & 1;          // M direction (2 waves)
  const int wn    = wave >> 1;         // N direction (4 waves)
  const int lrow  = lane & 15;
  const int khalf = lane >> 4;

  int KT = K >> 5;
  if (KCAP) {                          // P*V: only K-blocks at/below diagonal
    int cap = (m0 + 128) >> 5;
    if (cap < KT) KT = cap;
  }

  const __bf16* Abase = A  + (size_t)(m0 + srow) * lda + skoff;
  const __bf16* Bbase = Bm + (size_t)(n0 + srow) * ldb + skoff;

#if HAVE_ASYNC_LDS
  // --- async global->LDS staging: no VGPR round-trip, one barrier/K-step ---
  {
    async_cp16(Abase,     &As[0][srow][skoff]);
    async_cp16(Abase + 8, &As[0][srow][skoff + 8]);
    async_cp16(Bbase,     &Bs[0][srow][skoff]);
    async_cp16(Bbase + 8, &Bs[0][srow][skoff + 8]);
  }
  wait_asynccnt0();
  __syncthreads();
#else
  {
    bf16x8 a0 = ld8(Abase),     a1 = ld8(Abase + 8);
    bf16x8 b0 = ld8(Bbase),     b1 = ld8(Bbase + 8);
    *(bf16x8*)&As[0][srow][skoff]     = a0;
    *(bf16x8*)&As[0][srow][skoff + 8] = a1;
    *(bf16x8*)&Bs[0][srow][skoff]     = b0;
    *(bf16x8*)&Bs[0][srow][skoff + 8] = b1;
  }
  __syncthreads();
#endif

  v8f acc[4][2] = {};

  for (int kt = 0; kt < KT; ++kt) {
    const int  cur      = kt & 1;
    const bool has_next = (kt + 1) < KT;

#if HAVE_ASYNC_LDS
    if (has_next) {   // fill the other buffer; its readers passed last barrier
      const int ko = (kt + 1) << 5;
      const int nxt = 1 - cur;
      async_cp16(Abase + ko,     &As[nxt][srow][skoff]);
      async_cp16(Abase + ko + 8, &As[nxt][srow][skoff + 8]);
      async_cp16(Bbase + ko,     &Bs[nxt][srow][skoff]);
      async_cp16(Bbase + ko + 8, &Bs[nxt][srow][skoff + 8]);
    }
#else
    bf16x8 na0, na1, nb0, nb1;
    if (has_next) {
      const int ko = (kt + 1) << 5;
      na0 = ld8(Abase + ko);  na1 = ld8(Abase + ko + 8);
      nb0 = ld8(Bbase + ko);  nb1 = ld8(Bbase + ko + 8);
    }
#endif

    // A fragment: lanes 0-15 hold K{0..7,16..23}, lanes 16-31 K{8..15,24..31}
    v16bf afr[4];
#pragma unroll
    for (int mi = 0; mi < 4; ++mi) {
      const __bf16* p = &As[cur][wm * 64 + mi * 16 + lrow][khalf * 8];
      bf16x8 lo = ld8(p);
      bf16x8 hi = ld8(p + 16);
      afr[mi] = __builtin_shufflevector(lo, hi, 0,1,2,3,4,5,6,7,
                                                8,9,10,11,12,13,14,15);
    }
    // B fragment: lanes 0-15 hold K{0..7,16..23}, lanes 16-31 K{8..15,24..31}
    v16bf bfr[2];
#pragma unroll
    for (int ni = 0; ni < 2; ++ni) {
      const __bf16* p = &Bs[cur][wn * 32 + ni * 16 + lrow][khalf * 8];
      bf16x8 lo = ld8(p);
      bf16x8 hi = ld8(p + 16);
      bfr[ni] = __builtin_shufflevector(lo, hi, 0,1,2,3,4,5,6,7,
                                                8,9,10,11,12,13,14,15);
    }

#pragma unroll
    for (int mi = 0; mi < 4; ++mi)
#pragma unroll
      for (int ni = 0; ni < 2; ++ni)
        acc[mi][ni] = __builtin_amdgcn_wmma_f32_16x16x32_bf16(
            false, afr[mi], false, bfr[ni], (short)0, acc[mi][ni],
            false, false);

#if HAVE_ASYNC_LDS
    if (has_next) {
      wait_asynccnt0();                 // our async fills of buf nxt landed
      __syncthreads();                  // everyone's fills + reads of cur done
    }
#else
    if (has_next) {
      __syncthreads();
      const int nxt = 1 - cur;
      *(bf16x8*)&As[nxt][srow][skoff]     = na0;
      *(bf16x8*)&As[nxt][srow][skoff + 8] = na1;
      *(bf16x8*)&Bs[nxt][srow][skoff]     = nb0;
      *(bf16x8*)&Bs[nxt][srow][skoff + 8] = nb1;
      __syncthreads();
    }
#endif
  }

  // ---- epilogue -----------------------------------------------------------
  // D layout: lane = n + 16*(m/8) within tile; vgpr v -> row (m%8).
  float*  cf = (float*)Cout  + (size_t)bz * sC;
  __bf16* cb = (__bf16*)Cout + (size_t)bz * sC;
  const float* res = HAS_RES ? resid + (size_t)bz * sC : nullptr;

#pragma unroll
  for (int mi = 0; mi < 4; ++mi) {
#pragma unroll
    for (int ni = 0; ni < 2; ++ni) {
      const int n  = n0 + wn * 32 + ni * 16 + lrow;
      const int mb = m0 + wm * 64 + mi * 16 + khalf * 8;
#pragma unroll
      for (int v = 0; v < 8; ++v) {
        const int m = mb + v;
        float val = acc[mi][ni][v] * scale;
        if (HAS_BIAS) val += bias[n];
        if (RELU)     val = fmaxf(val, 0.0f);
        if (HAS_RES)  val += res[(size_t)m * ldc + n];
        if (TRANS_OUT) {
          // V^T store: token m = (b,t); layout Vt[b][C=1024][T=2048]
          const int b = m >> 11, t = m & 2047;
          cb[((size_t)b * 1024 + n) * 2048 + t] = f2bf(val);
        } else if (OUT_BF16) {
          cb[(size_t)m * ldc + n] = f2bf(val);
        } else {
          cf[(size_t)m * ldc + n] = val;
        }
      }
    }
  }
}

// ---------------------------------------------------------------------------
// Row LayerNorm (C=1024) -> bf16.  One 256-thread WG per row, 4 elems/thread.
// ---------------------------------------------------------------------------
__global__ __launch_bounds__(256)
void ln_bf16_kernel(const float* __restrict__ x, const float* __restrict__ g,
                    const float* __restrict__ beta, __bf16* __restrict__ out)
{
  const int row = blockIdx.x;
  const int tid = threadIdx.x;
  const float* xr = x + (size_t)row * 1024;

  float4 xv = *(const float4*)(xr + tid * 4);
  float s = xv.x + xv.y + xv.z + xv.w;
  float q = xv.x * xv.x + xv.y * xv.y + xv.z * xv.z + xv.w * xv.w;

  __shared__ float sh_s[256], sh_q[256];
  sh_s[tid] = s; sh_q[tid] = q;
  __syncthreads();
  for (int st = 128; st > 0; st >>= 1) {
    if (tid < st) { sh_s[tid] += sh_s[tid + st]; sh_q[tid] += sh_q[tid + st]; }
    __syncthreads();
  }
  const float mean = sh_s[0] * (1.0f / 1024.0f);
  const float var  = sh_q[0] * (1.0f / 1024.0f) - mean * mean;
  const float rstd = rsqrtf(var + 1e-5f);

  const int c = tid * 4;
  bf16x4 o;
  o.x = f2bf((xv.x - mean) * rstd * g[c + 0] + beta[c + 0]);
  o.y = f2bf((xv.y - mean) * rstd * g[c + 1] + beta[c + 1]);
  o.z = f2bf((xv.z - mean) * rstd * g[c + 2] + beta[c + 2]);
  o.w = f2bf((xv.w - mean) * rstd * g[c + 3] + beta[c + 3]);
  *(bf16x4*)(out + (size_t)row * 1024 + c) = o;
}

// ---------------------------------------------------------------------------
// Causal softmax: row (b,t) reads scores s<=t (rest is never-written garbage,
// never touched), writes full 2048-wide bf16 row (zeros past the diagonal).
// ---------------------------------------------------------------------------
__global__ __launch_bounds__(256)
void softmax_causal(const float* __restrict__ S, __bf16* __restrict__ P)
{
  const int r = blockIdx.x;
  const int b = r >> 11, t = r & 2047;
  const float* srow = S + ((size_t)b * 2048 + t) * 2048;
  __bf16*      prow = P + ((size_t)b * 2048 + t) * 2048;
  const int tid = threadIdx.x;

  __shared__ float red[256];

  float mx = -3.4e38f;
  for (int j = tid; j <= t; j += 256) mx = fmaxf(mx, srow[j]);
  red[tid] = mx; __syncthreads();
  for (int st = 128; st > 0; st >>= 1) {
    if (tid < st) red[tid] = fmaxf(red[tid], red[tid + st]);
    __syncthreads();
  }
  const float gm = red[0];
  __syncthreads();

  float sum = 0.0f;
  for (int j = tid; j <= t; j += 256) sum += __expf(srow[j] - gm);
  red[tid] = sum; __syncthreads();
  for (int st = 128; st > 0; st >>= 1) {
    if (tid < st) red[tid] += red[tid + st];
    __syncthreads();
  }
  const float inv = 1.0f / red[0];

  for (int j = tid; j < 2048; j += 256) {
    const float pv = (j <= t) ? __expf(srow[j] - gm) * inv : 0.0f;
    prow[j] = f2bf(pv);
  }
}

// ---------------------------------------------------------------------------
__global__ __launch_bounds__(256)
void f32_to_bf16(const float* __restrict__ in, __bf16* __restrict__ out, int n)
{
  const int i = blockIdx.x * 256 + threadIdx.x;
  if (i < n) out[i] = f2bf(in[i]);
}

// ---------------------------------------------------------------------------
extern "C" void kernel_launch(void* const* d_in, const int* in_sizes, int n_in,
                              void* d_out, int out_size, void* d_ws, size_t ws_size,
                              hipStream_t stream)
{
  (void)in_sizes; (void)n_in; (void)out_size; (void)ws_size;
  const float* inputs = (const float*)d_in[0];
  const float* Wq  = (const float*)d_in[1];
  const float* Wk  = (const float*)d_in[2];
  const float* Wv  = (const float*)d_in[3];
  const float* W1  = (const float*)d_in[4];
  const float* b1  = (const float*)d_in[5];
  const float* W2  = (const float*)d_in[6];
  const float* b2  = (const float*)d_in[7];
  const float* g1  = (const float*)d_in[8];
  const float* be1 = (const float*)d_in[9];
  const float* g2  = (const float*)d_in[10];
  const float* be2 = (const float*)d_in[11];
  float* out = (float*)d_out;

  // workspace layout (bytes)
  char* ws = (char*)d_ws;
  const size_t MB = 1ull << 20;
  __bf16* WqB = (__bf16*)(ws + 0 * MB);     // 2 MB
  __bf16* WkB = (__bf16*)(ws + 2 * MB);     // 2 MB
  __bf16* WvB = (__bf16*)(ws + 4 * MB);     // 2 MB
  __bf16* W1B = (__bf16*)(ws + 6 * MB);     // 8 MB
  __bf16* W2B = (__bf16*)(ws + 14 * MB);    // 8 MB
  __bf16* xn  = (__bf16*)(ws + 22 * MB);    // 32 MB (reused as h2 after QKV)
  __bf16* Qb  = (__bf16*)(ws + 54 * MB);    // 32 MB
  __bf16* Kb  = (__bf16*)(ws + 86 * MB);    // 32 MB
  __bf16* Vt  = (__bf16*)(ws + 118 * MB);   // 32 MB (V transposed per batch)
  __bf16* P   = (__bf16*)(ws + 150 * MB);   // 64 MB
  float*  S   = (float*)(ws + 214 * MB);    // 128 MB (reused as H1 bf16)
  __bf16* H1  = (__bf16*)(ws + 214 * MB);

  // 1) weights -> bf16
  f32_to_bf16<<<4096, 256, 0, stream>>>(Wq, WqB, 1024 * 1024);
  f32_to_bf16<<<4096, 256, 0, stream>>>(Wk, WkB, 1024 * 1024);
  f32_to_bf16<<<4096, 256, 0, stream>>>(Wv, WvB, 1024 * 1024);
  f32_to_bf16<<<16384, 256, 0, stream>>>(W1, W1B, 4096 * 1024);
  f32_to_bf16<<<16384, 256, 0, stream>>>(W2, W2B, 4096 * 1024);

  // 2) LN1
  ln_bf16_kernel<<<16384, 256, 0, stream>>>(inputs, g1, be1, xn);

  // 3-5) Q (scale 1/sqrt(C) folded in), K, V (stored transposed)
  gemm_nt<true, false, false, false, false, false, false>
      <<<dim3(8, 128, 1), 256, 0, stream>>>(
      xn, 1024, 0, WqB, 1024, 0, Qb, 1024, 0, nullptr, nullptr,
      1.0f / 32.0f, 16384, 1024, 1024);
  gemm_nt<true, false, false, false, false, false, false>
      <<<dim3(8, 128, 1), 256, 0, stream>>>(
      xn, 1024, 0, WkB, 1024, 0, Kb, 1024, 0, nullptr, nullptr,
      1.0f, 16384, 1024, 1024);
  gemm_nt<true, true, false, false, false, false, false>
      <<<dim3(8, 128, 1), 256, 0, stream>>>(
      xn, 1024, 0, WvB, 1024, 0, Vt, 2048, 0, nullptr, nullptr,
      1.0f, 16384, 1024, 1024);

  // 6) scores = Q K^T (batched, skip fully-masked tiles)
  gemm_nt<false, false, false, true, false, false, false>
      <<<dim3(16, 16, 8), 256, 0, stream>>>(
      Qb, 1024, (long)2048 * 1024, Kb, 1024, (long)2048 * 1024,
      S, 2048, (long)2048 * 2048, nullptr, nullptr,
      1.0f, 2048, 2048, 1024);

  // 7) causal softmax -> P (bf16)
  softmax_causal<<<16384, 256, 0, stream>>>(S, P);

  // 8) new_v = P V + inputs  (K-loop capped at diagonal) -> d_out (f32)
  gemm_nt<false, false, false, false, true, false, true>
      <<<dim3(8, 16, 8), 256, 0, stream>>>(
      P, 2048, (long)2048 * 2048, Vt, 2048, (long)1024 * 2048,
      out, 1024, (long)2048 * 1024, nullptr, inputs,
      1.0f, 2048, 1024, 2048);

  // 9) LN2 (reuse xn buffer as h)
  ln_bf16_kernel<<<16384, 256, 0, stream>>>(out, g2, be2, xn);

  // 10) H1 = relu(h W1^T + b1) -> bf16 (reuses S memory)
  gemm_nt<true, false, true, false, false, true, false>
      <<<dim3(32, 128, 1), 256, 0, stream>>>(
      xn, 1024, 0, W1B, 1024, 0, H1, 4096, 0, b1, nullptr,
      1.0f, 16384, 4096, 1024);

  // 11) out = H1 W2^T + b2 + new_v  (in-place residual from d_out)
  gemm_nt<false, false, false, false, false, true, true>
      <<<dim3(8, 128, 1), 256, 0, stream>>>(
      H1, 4096, 0, W2B, 4096, 0, out, 1024, 0, b2, (const float*)out,
      1.0f, 16384, 1024, 4096);
}
